// EncoderBlock_14680198218343
// MI455X (gfx1250) — matrix operand order repaired and checked by reading, not compiled, not run on hardware
//
#include <hip/hip_runtime.h>

// ---------------- constants (match reference) ----------------
constexpr int Bc   = 8;
constexpr int Sc   = 1024;
constexpr int DIMc = 1024;
constexpr int Hc   = 16;
constexpr int DHc  = 64;
constexpr int HIDc = 512;
constexpr int Mc   = Bc * Sc;      // 8192 rows

// ---------------- types ----------------
typedef __bf16 v16bf __attribute__((ext_vector_type(16)));
typedef __bf16 v8bf  __attribute__((ext_vector_type(8)));
typedef float  v8f   __attribute__((ext_vector_type(8)));
typedef unsigned int v4u  __attribute__((ext_vector_type(4)));
typedef int          v8i_ __attribute__((ext_vector_type(8)));
typedef int          v4i_ __attribute__((ext_vector_type(4)));

__device__ __forceinline__ unsigned short f2bf(float f) {
  union { float f; unsigned u; } v; v.f = f;
  unsigned r = v.u + 0x7FFFu + ((v.u >> 16) & 1u);
  return (unsigned short)(r >> 16);
}

__device__ __forceinline__ v16bf cat8(v8bf lo, v8bf hi) {
  return __builtin_shufflevector(lo, hi, 0,1,2,3,4,5,6,7,8,9,10,11,12,13,14,15);
}

// XOR-butterfly lane exchange via ds_swizzle (imm group-of-32 pattern: and=0x1f, xor=mask)
template <int IMM>
__device__ __forceinline__ float swz(float v) {
  return __builtin_bit_cast(
      float, __builtin_amdgcn_ds_swizzle(__builtin_bit_cast(int, v), IMM));
}
__device__ __forceinline__ float grp16_max(float v) {
  v = fmaxf(v, swz<0x041F>(v));
  v = fmaxf(v, swz<0x081F>(v));
  v = fmaxf(v, swz<0x101F>(v));
  v = fmaxf(v, swz<0x201F>(v));
  return v;
}
__device__ __forceinline__ float grp16_sum(float v) {
  v += swz<0x041F>(v);
  v += swz<0x081F>(v);
  v += swz<0x101F>(v);
  v += swz<0x201F>(v);
  return v;
}

// ---------------- TDM: stage a 2D bf16 panel (rows x dim0) into LDS ----------------
// D# per CDNA5 ISA ch.8: group0 = {flags/count, lds_addr, global_addr lo/hi+type},
// group1 = {mask/data_size, dims, tile dims, strides}. data_size=1 (2 bytes).
// This toolchain exposes the 6-arg builtin: (g0, g1, g2, g3, g4, cpol);
// groups 2..4 are zero-filled for a 2D tensor (VADDR2/3 NULL-equivalent).
__device__ __forceinline__ void tdm_load_panel_2d(unsigned lds_byte_addr,
                                                  const unsigned short* gptr,
                                                  unsigned dim0_elems,
                                                  unsigned rows,
                                                  unsigned stride0_elems) {
  unsigned long long ga = (unsigned long long)(const void*)gptr;
  v4u g0;
  g0[0] = 1u;                                    // count=1, user mode, no gather
  g0[1] = lds_byte_addr;                         // lds_addr
  g0[2] = (unsigned)(ga & 0xFFFFFFFFu);          // global_addr[31:0]
  g0[3] = (unsigned)((ga >> 32) & 0x01FFFFFFu)   // global_addr[56:32]
          | (2u << 30);                          // type = 2 ("image")
  v8i_ g1;
  g1[0] = (int)(1u << 16);                       // workgroup_mask=0, data_size=1 (2B)
  g1[1] = (int)((dim0_elems & 0xFFFFu) << 16);   // tensor_dim0[15:0]
  g1[2] = (int)(((dim0_elems >> 16) & 0xFFFFu)   // tensor_dim0[31:16]
          | ((rows & 0xFFFFu) << 16));           // tensor_dim1[15:0]
  g1[3] = (int)(((rows >> 16) & 0xFFFFu)         // tensor_dim1[31:16]
          | ((dim0_elems & 0xFFFFu) << 16));     // tile_dim0 = dim0
  g1[4] = (int)(rows & 0xFFFFu);                 // tile_dim1 = rows, tile_dim2 = 0
  g1[5] = (int)stride0_elems;                    // tensor_dim0_stride[31:0]
  g1[6] = 0;                                     // stride0[47:32], stride1[15:0]
  g1[7] = 0;                                     // stride1[47:16]
  v4i_ z4 = {0, 0, 0, 0};
  v8i_ z8 = {0, 0, 0, 0, 0, 0, 0, 0};
  __builtin_amdgcn_tensor_load_to_lds(g0, g1, z4, z4, z8, 0);
  __builtin_amdgcn_s_wait_tensorcnt(0);
}

// ---------------- conversion kernels ----------------
__global__ void f32_to_bf16_kernel(const float* __restrict__ in,
                                   unsigned short* __restrict__ out, int n) {
  int i = blockIdx.x * 256 + threadIdx.x;
  if (i < n) out[i] = f2bf(in[i]);
}

// W[K,N] fp32 row-major -> bf16 WMMA-B fragment-major (32x16 tiles, 512 contiguous)
__global__ void wfrag_kernel(const float* __restrict__ W,
                             unsigned short* __restrict__ out, int K, int N) {
  int i = blockIdx.x * 256 + threadIdx.x;
  if (i >= K * N) return;
  int k = i / N, n = i % N;
  int kt = k >> 5, ko = k & 31, kh = ko >> 4, kl = ko & 15;
  int nt = n >> 4, nn = n & 15;
  size_t dst = ((((size_t)nt * (size_t)(K >> 5) + kt) << 5) + (kh << 4) + nn) * 16 + kl;
  out[dst] = f2bf(W[i]);
}

// ---------------- GEMM: C[M,N] = A[M,K](bf16 rowmajor) @ Bfrag ----------------
// All 8 waves of a block share one 16-row A panel; it is staged into LDS once
// via the Tensor Data Mover, then the K-loop reads A fragments with ds_load.
// MODE 0: f32 row-major out (+bias if non-null)
// MODE 1: bf16 out, scatter to Q/K layout [B][H][S][DH]
// MODE 2: bf16 out, scatter to V^T layout [B][H][DH][S]
// MODE 3: bf16 out, relu(x+bias), row-major
template <int MODE>
__global__ __launch_bounds__(256) void gemm_bf16(
    const unsigned short* __restrict__ A,
    const unsigned short* __restrict__ Bf,
    void* __restrict__ Out,
    const float* __restrict__ bias,
    int M, int N, int K) {
  __shared__ __attribute__((aligned(128))) unsigned short Apanel[16 * 1024];
  const int lane = threadIdx.x & 31;
  const int wave = threadIdx.x >> 5;
  const int wid  = blockIdx.x * (blockDim.x >> 5) + wave;
  const int ngrp = N >> 6;            // 64-column groups (multiple of 8)
  const int mt   = wid / ngrp;        // identical for all 8 waves in a block
  const int ng   = wid % ngrp;
  const int m0   = mt << 4;
  const int ktiles = K >> 5;
  const int half = lane >> 4, r = lane & 15;

  // ---- TDM: stage A panel (16 x K bf16) into LDS, one load per block ----
  if (wave == 0) {
    tdm_load_panel_2d((unsigned)(size_t)&Apanel[0], A + (size_t)m0 * (size_t)K,
                      (unsigned)K, 16u, (unsigned)K);
  }
  __syncthreads();

  v8f acc[4];
#pragma unroll
  for (int t = 0; t < 4; ++t)
#pragma unroll
    for (int i = 0; i < 8; ++i) acc[t][i] = 0.f;

  const unsigned short* ap = &Apanel[r * K + half * 16];
  for (int kt = 0; kt < ktiles; ++kt) {
    v8bf lo = *(const v8bf*)(ap);
    v8bf hi = *(const v8bf*)(ap + 8);
    v16bf af = cat8(lo, hi);
    ap += 32;
#pragma unroll
    for (int t = 0; t < 4; ++t) {
      const unsigned short* bp =
          Bf + (((size_t)(ng * 4 + t) * (size_t)ktiles + kt) << 9) + (lane << 4);
      v16bf bfr = *(const v16bf*)bp;
      acc[t] = __builtin_amdgcn_wmma_f32_16x16x32_bf16(
          false, af, false, bfr, (short)0, acc[t], false, false);
    }
  }

  // epilogue
#pragma unroll
  for (int t = 0; t < 4; ++t) {
    int col = (ng << 6) + t * 16 + r;
#pragma unroll
    for (int j = 0; j < 8; ++j) {
      int row = m0 + half * 8 + j;
      float v = acc[t][j];
      if (MODE == 0) {
        float bv = bias ? bias[col] : 0.f;
        ((float*)Out)[(size_t)row * (size_t)N + col] = v + bv;
      } else if (MODE == 1) {
        int b = row >> 10, q = row & (Sc - 1);
        int h = col >> 6, dh = col & (DHc - 1);
        ((unsigned short*)Out)[((((size_t)b * Hc + h) << 10) + q) * DHc + dh] = f2bf(v);
      } else if (MODE == 2) {
        int b = row >> 10, q = row & (Sc - 1);
        int h = col >> 6, dh = col & (DHc - 1);
        ((unsigned short*)Out)[((((size_t)b * Hc + h) * DHc + dh) << 10) + q] = f2bf(v);
      } else {
        float x = v + bias[col];
        ((unsigned short*)Out)[(size_t)row * (size_t)N + col] = f2bf(x > 0.f ? x : 0.f);
      }
    }
  }
}

// ---------------- fused flash attention ----------------
// grid (S/16/8, B*H), block 256 (8 waves); one wave per 16-query tile.
__global__ __launch_bounds__(256) void attn_kernel(
    const unsigned short* __restrict__ Q,
    const unsigned short* __restrict__ Kb,
    const unsigned short* __restrict__ Vt,
    unsigned short* __restrict__ ctx) {
  __shared__ __attribute__((aligned(32))) unsigned short plds[8 * 512];
  const int lane = threadIdx.x & 31;
  const int wave = threadIdx.x >> 5;
  const int qt   = blockIdx.x * 8 + wave;
  const int bh   = blockIdx.y;
  const int b    = bh >> 4, h = bh & 15;
  const int q0   = qt << 4;
  const int half = lane >> 4, r = lane & 15;

  const unsigned short* Qh = Q  + ((size_t)bh << 16);  // bh * S*DH
  const unsigned short* Kh = Kb + ((size_t)bh << 16);
  const unsigned short* Vh = Vt + ((size_t)bh << 16);
  unsigned short* pl = plds + wave * 512;

  // Q fragments for k0 = 0, 32 (DH = 64)
  v16bf qa[2];
  {
    const unsigned short* qrow = Qh + (size_t)(q0 + r) * DHc + half * 16;
#pragma unroll
    for (int kq = 0; kq < 2; ++kq) {
      v8bf lo = *(const v8bf*)(qrow + kq * 32);
      v8bf hi = *(const v8bf*)(qrow + kq * 32 + 8);
      qa[kq] = cat8(lo, hi);
    }
  }

  float mrow[8], lrow[8];
  v8f acc[4];
#pragma unroll
  for (int j = 0; j < 8; ++j) { mrow[j] = -1e30f; lrow[j] = 0.f; }
#pragma unroll
  for (int t = 0; t < 4; ++t)
#pragma unroll
    for (int i = 0; i < 8; ++i) acc[t][i] = 0.f;

  for (int kb = 0; kb < Sc; kb += 32) {
    // ---- two 16x16 score tiles: S = Q @ K^T ----
    v8f sc[2];
#pragma unroll
    for (int t = 0; t < 2; ++t) {
#pragma unroll
      for (int i = 0; i < 8; ++i) sc[t][i] = 0.f;
      const unsigned short* kp = Kh + (size_t)(kb + t * 16 + r) * DHc + half * 16;
#pragma unroll
      for (int kq = 0; kq < 2; ++kq) {
        v16bf kf = *(const v16bf*)(kp + kq * 32);
        sc[t] = __builtin_amdgcn_wmma_f32_16x16x32_bf16(
            false, qa[kq], false, kf, (short)0, sc[t], false, false);
      }
    }

    // ---- online softmax update (row j lives in 16 lanes of VGPR j) ----
    float corr[8];
#pragma unroll
    for (int j = 0; j < 8; ++j) {
      float s0 = sc[0][j] * 0.125f;   // 1/sqrt(DH)
      float s1 = sc[1][j] * 0.125f;
      float mx = grp16_max(fmaxf(s0, s1));
      float mnew = fmaxf(mrow[j], mx);
      float c  = __expf(mrow[j] - mnew);
      float p0 = __expf(s0 - mnew);
      float p1 = __expf(s1 - mnew);
      lrow[j] = lrow[j] * c + grp16_sum(p0 + p1);
      mrow[j] = mnew;
      corr[j] = c;
      // C-layout -> LDS row-major P tile 16x32 (bf16)
      pl[(j + half * 8) * 32 + r]      = f2bf(p0);
      pl[(j + half * 8) * 32 + 16 + r] = f2bf(p1);
    }
    asm volatile("s_wait_dscnt 0" ::: "memory");

    // ---- reload P in WMMA-A layout ----
    v16bf pa;
    {
      const unsigned short* pp = pl + r * 32 + half * 16;
      v8bf lo = *(const v8bf*)pp;
      v8bf hi = *(const v8bf*)(pp + 8);
      pa = cat8(lo, hi);
    }

    // ---- rescale accumulators and add P @ V (V stored transposed) ----
#pragma unroll
    for (int t = 0; t < 4; ++t) {
#pragma unroll
      for (int j = 0; j < 8; ++j) acc[t][j] *= corr[j];
      const unsigned short* vp = Vh + (size_t)(t * 16 + r) * Sc + kb + half * 16;
      v16bf vf = *(const v16bf*)vp;
      acc[t] = __builtin_amdgcn_wmma_f32_16x16x32_bf16(
          false, pa, false, vf, (short)0, acc[t], false, false);
    }
  }

  // ---- finalize: ctx[b, q, h*64+dh] (bf16 row-major for the Wo GEMM) ----
#pragma unroll
  for (int j = 0; j < 8; ++j) {
    float inv = 1.0f / lrow[j];
    int row = q0 + half * 8 + j;
    size_t base = ((size_t)b * Sc + row) * DIMc + h * DHc;
#pragma unroll
    for (int t = 0; t < 4; ++t)
      ctx[base + t * 16 + r] = f2bf(acc[t][j] * inv);
  }
}

// ---------------- fused residual-add + LayerNorm ----------------
// one block (256 threads) per row of 1024
__global__ __launch_bounds__(256) void ln_kernel(
    const float* __restrict__ a, const float* __restrict__ res,
    const float* __restrict__ g, const float* __restrict__ beta,
    float* __restrict__ out_f32, unsigned short* __restrict__ out_bf) {
  const int row = blockIdx.x;
  const int tid = threadIdx.x;
  __shared__ float sred[16];
  float vals[4];
  float s = 0.f, sq = 0.f;
#pragma unroll
  for (int i = 0; i < 4; ++i) {
    int c = tid + i * 256;
    float v = a[(size_t)row * DIMc + c] + res[(size_t)row * DIMc + c];
    vals[i] = v; s += v; sq += v * v;
  }
  // wave (32-lane) XOR reduction via ds_swizzle
  s = s + swz<0x041F>(s); sq = sq + swz<0x041F>(sq);
  s = s + swz<0x081F>(s); sq = sq + swz<0x081F>(sq);
  s = s + swz<0x101F>(s); sq = sq + swz<0x101F>(sq);
  s = s + swz<0x201F>(s); sq = sq + swz<0x201F>(sq);
  s = s + swz<0x401F>(s); sq = sq + swz<0x401F>(sq);
  if ((tid & 31) == 0) { sred[tid >> 5] = s; sred[8 + (tid >> 5)] = sq; }
  __syncthreads();
  if (tid == 0) {
    float ts = 0.f, tq = 0.f;
    for (int i = 0; i < 8; ++i) { ts += sred[i]; tq += sred[8 + i]; }
    sred[0] = ts; sred[8] = tq;
  }
  __syncthreads();
  float mu   = sred[0] * (1.f / DIMc);
  float var  = sred[8] * (1.f / DIMc) - mu * mu;
  float rstd = rsqrtf(var + 1e-5f);
#pragma unroll
  for (int i = 0; i < 4; ++i) {
    int c = tid + i * 256;
    float o = (vals[i] - mu) * rstd * g[c] + beta[c];
    out_f32[(size_t)row * DIMc + c] = o;
    if (out_bf) out_bf[(size_t)row * DIMc + c] = f2bf(o);
  }
}

// ---------------- host orchestration ----------------
extern "C" void kernel_launch(void* const* d_in, const int* in_sizes, int n_in,
                              void* d_out, int out_size, void* d_ws, size_t ws_size,
                              hipStream_t stream) {
  const float* x     = (const float*)d_in[0];
  const float* Wq    = (const float*)d_in[1];
  const float* Wk    = (const float*)d_in[2];
  const float* Wv    = (const float*)d_in[3];
  const float* Wo    = (const float*)d_in[4];
  const float* ln1_g = (const float*)d_in[5];
  const float* ln1_b = (const float*)d_in[6];
  const float* W1    = (const float*)d_in[7];
  const float* b1    = (const float*)d_in[8];
  const float* W2    = (const float*)d_in[9];
  const float* b2    = (const float*)d_in[10];
  const float* ln2_g = (const float*)d_in[11];
  const float* ln2_b = (const float*)d_in[12];
  float* out = (float*)d_out;

  char* ws = (char*)d_ws;
  auto alloc = [&](size_t bytes) {
    char* p = ws;
    ws += (bytes + 255) & ~(size_t)255;
    return p;
  };
  unsigned short* Xb    = (unsigned short*)alloc((size_t)Mc * DIMc * 2);
  unsigned short* Wqb   = (unsigned short*)alloc((size_t)DIMc * DIMc * 2);
  unsigned short* Wkb   = (unsigned short*)alloc((size_t)DIMc * DIMc * 2);
  unsigned short* Wvb   = (unsigned short*)alloc((size_t)DIMc * DIMc * 2);
  unsigned short* Wob   = (unsigned short*)alloc((size_t)DIMc * DIMc * 2);
  unsigned short* W1b   = (unsigned short*)alloc((size_t)DIMc * HIDc * 2);
  unsigned short* W2b   = (unsigned short*)alloc((size_t)HIDc * DIMc * 2);
  unsigned short* Qb    = (unsigned short*)alloc((size_t)Mc * DIMc * 2);
  unsigned short* Kb    = (unsigned short*)alloc((size_t)Mc * DIMc * 2);
  unsigned short* Vtb   = (unsigned short*)alloc((size_t)Mc * DIMc * 2);
  unsigned short* ctxb  = (unsigned short*)alloc((size_t)Mc * DIMc * 2);
  float*          mha   = (float*)alloc((size_t)Mc * DIMc * 4);
  float*          hf32  = (float*)alloc((size_t)Mc * DIMc * 4);
  unsigned short* hbf   = (unsigned short*)alloc((size_t)Mc * DIMc * 2);
  unsigned short* ff1b  = (unsigned short*)alloc((size_t)Mc * HIDc * 2);
  float*          ff2   = (float*)alloc((size_t)Mc * DIMc * 4);

  // 1) convert activations / weights to bf16 (weights -> fragment-major)
  {
    int n = Mc * DIMc;
    f32_to_bf16_kernel<<<(n + 255) / 256, 256, 0, stream>>>(x, Xb, n);
    int nw = DIMc * DIMc;
    wfrag_kernel<<<(nw + 255) / 256, 256, 0, stream>>>(Wq, Wqb, DIMc, DIMc);
    wfrag_kernel<<<(nw + 255) / 256, 256, 0, stream>>>(Wk, Wkb, DIMc, DIMc);
    wfrag_kernel<<<(nw + 255) / 256, 256, 0, stream>>>(Wv, Wvb, DIMc, DIMc);
    wfrag_kernel<<<(nw + 255) / 256, 256, 0, stream>>>(Wo, Wob, DIMc, DIMc);
    int n1 = DIMc * HIDc;
    wfrag_kernel<<<(n1 + 255) / 256, 256, 0, stream>>>(W1, W1b, DIMc, HIDc);
    wfrag_kernel<<<(n1 + 255) / 256, 256, 0, stream>>>(W2, W2b, HIDc, DIMc);
  }

  // 2) Q/K/V projections (scatter to attention-friendly layouts)
  {
    int blocks = (Mc / 16) * (DIMc / 64) / 8;
    gemm_bf16<1><<<blocks, 256, 0, stream>>>(Xb, Wqb, Qb,  nullptr, Mc, DIMc, DIMc);
    gemm_bf16<1><<<blocks, 256, 0, stream>>>(Xb, Wkb, Kb,  nullptr, Mc, DIMc, DIMc);
    gemm_bf16<2><<<blocks, 256, 0, stream>>>(Xb, Wvb, Vtb, nullptr, Mc, DIMc, DIMc);
  }

  // 3) fused flash attention -> ctx (bf16 row-major)
  attn_kernel<<<dim3(Sc / 16 / 8, Bc * Hc), 256, 0, stream>>>(Qb, Kb, Vtb, ctxb);

  // 4) output projection
  {
    int blocks = (Mc / 16) * (DIMc / 64) / 8;
    gemm_bf16<0><<<blocks, 256, 0, stream>>>(ctxb, Wob, mha, nullptr, Mc, DIMc, DIMc);
  }

  // 5) residual + LN1 -> h (f32 + bf16)
  ln_kernel<<<Mc, 256, 0, stream>>>(mha, x, ln1_g, ln1_b, hf32, hbf);

  // 6) FFN
  {
    int blocks1 = (Mc / 16) * (HIDc / 64) / 8;
    gemm_bf16<3><<<blocks1, 256, 0, stream>>>(hbf, W1b, ff1b, b1, Mc, HIDc, DIMc);
    int blocks2 = (Mc / 16) * (DIMc / 64) / 8;
    gemm_bf16<0><<<blocks2, 256, 0, stream>>>(ff1b, W2b, ff2, b2, Mc, DIMc, HIDc);
  }

  // 7) residual + LN2 -> out
  ln_kernel<<<Mc, 256, 0, stream>>>(ff2, hf32, ln2_g, ln2_b, out, nullptr);
}
